// EnergyAttention_81011673137211
// MI455X (gfx1250) — matrix operand order repaired and checked by reading, not compiled, not run on hardware
//
#include <hip/hip_runtime.h>

typedef __attribute__((ext_vector_type(16))) _Float16 v16h;
typedef __attribute__((ext_vector_type(8)))  _Float16 v8h;
typedef __attribute__((ext_vector_type(4)))  _Float16 v4h;
typedef __attribute__((ext_vector_type(8)))  float    v8f;
typedef __attribute__((ext_vector_type(4)))  float    v4f;

#define NB 2
#define NH 16
#define NS 1024
#define NZ 64
#define ND 1024
#define NHZ (NH * NZ)
#define BETA_F 0.125f
#define POS_SCALE_F 0.001f

#define TM 64
#define TN 64
#define TK 32
#define NTHR 128

union U16x16 {
  v16h v;
  v8h  h[2];
};

// CDNA5 async global->LDS copy (ASYNCcnt-tracked), GVS addressing:
// saddr(SGPR pair) + 32-bit per-lane byte offset; dst VGPR = LDS byte offset.
#define ASYNC_LDS_B128(ldsoff, goff, gbase)                                  \
  asm volatile("global_load_async_to_lds_b128 %0, %1, %2"                    \
               ::"v"(ldsoff), "v"(goff), "s"(gbase)                          \
               : "memory")
#define WAIT_ASYNC() asm volatile("s_wait_asynccnt 0" ::: "memory")

__device__ __forceinline__ unsigned lds_off(const void* p) {
  return (unsigned)(unsigned long long)p;
}

// ---------------------------------------------------------------------------
// f32 -> f16 elementwise convert (4 elements / thread)
// ---------------------------------------------------------------------------
__global__ __launch_bounds__(256) void cvt_f32_f16(const float* __restrict__ in,
                                                   _Float16* __restrict__ out,
                                                   int n4) {
  int i = blockIdx.x * 256 + threadIdx.x;
  if (i < n4) {
    v4f x = *(const v4f*)(in + 4 * i);
    v4h y;
    y[0] = (_Float16)x[0];
    y[1] = (_Float16)x[1];
    y[2] = (_Float16)x[2];
    y[3] = (_Float16)x[3];
    *(v4h*)(out + 4 * i) = y;
  }
}

// ---------------------------------------------------------------------------
// Batched f16 GEMM, compile-time specialized, LDS double-buffered.
//   C[m,n] = alpha * sum_k A[m,k]*B[k,n], f32 accumulate via WMMA.
// k-contiguous operands stream via async global->LDS (ASYNCcnt);
// transposed operands use register-prefetch + post-compute LDS scatter.
// Workgroup = 128 threads = 4 wave32; each wave owns 32x32 (2x2 WMMA tiles).
// ---------------------------------------------------------------------------
template <int K, int As_m, int As_k, int Bs_k, int Bs_n, int Cs_m, int Cs_n,
          int Ab, int Ah, int Bb, int Bh, int Cb, int Ch, int HDIV>
__global__ __launch_bounds__(NTHR) void gemm_f16(
    const _Float16* __restrict__ A, const _Float16* __restrict__ B,
    _Float16* __restrict__ C, float alpha) {
  __shared__ alignas(16) _Float16 Asm[2][TM][TK];
  __shared__ alignas(16) _Float16 Bsm[2][TN][TK];  // [n][k]

  const int z = blockIdx.z;
  const int b = z / HDIV;
  const int h = z % HDIV;
  A += b * Ab + h * Ah;
  B += b * Bb + h * Bh;
  C += b * Cb + h * Ch;

  const int tileM = blockIdx.y * TM;
  const int tileN = blockIdx.x * TN;
  const int tid   = threadIdx.x;
  const int wave  = tid >> 5;
  const int lane  = tid & 31;
  const int wm    = (wave >> 1) * 32;
  const int wn    = (wave & 1) * 32;
  const int rl    = lane & 15;
  const int hs    = lane >> 4;   // half-select
  const int abase = hs * 8;      // A-frag K base within 32
  const int bbase = hs * 8;      // B-frag K base within 32

  // per-thread chunk coordinates (2 x 16B chunks per tile per thread)
  const int acr[2] = {tid >> 2, (tid + NTHR) >> 2};          // k-contig A row
  const int ack    = (tid & 3) << 3;                         // k-contig A col
  const int tkk[2] = {tid >> 3, (tid + NTHR) >> 3};          // transpose k idx
  const int tc8    = (tid & 7) << 3;                         // transpose m/n8

  v8f acc[2][2] = {};
  v8h pa[2], pb[2];  // prefetch registers for transposed operands

  // ---- stage tile 0 into buffer 0 ----
  if constexpr (As_k == 1) {
#pragma unroll
    for (int u = 0; u < 2; ++u)
      ASYNC_LDS_B128(lds_off(&Asm[0][acr[u]][ack]),
                     (unsigned)(((tileM + acr[u]) * As_m + ack) * 2), A);
  } else {
#pragma unroll
    for (int u = 0; u < 2; ++u) {
      v8h v = *(const v8h*)&A[tkk[u] * As_k + (tileM + tc8)];
#pragma unroll
      for (int j = 0; j < 8; ++j) Asm[0][tc8 + j][tkk[u]] = v[j];
    }
  }
  if constexpr (Bs_k == 1) {
#pragma unroll
    for (int u = 0; u < 2; ++u)
      ASYNC_LDS_B128(lds_off(&Bsm[0][acr[u]][ack]),
                     (unsigned)(((tileN + acr[u]) * Bs_n + ack) * 2), B);
  } else {
#pragma unroll
    for (int u = 0; u < 2; ++u) {
      v8h v = *(const v8h*)&B[tkk[u] * Bs_k + (tileN + tc8)];
#pragma unroll
      for (int j = 0; j < 8; ++j) Bsm[0][tc8 + j][tkk[u]] = v[j];
    }
  }
  if constexpr (As_k == 1 || Bs_k == 1) WAIT_ASYNC();
  __syncthreads();

  int cur = 0;
  for (int k0 = 0; k0 < K; k0 += TK) {
    const int nxt      = cur ^ 1;
    const bool hasNext = (k0 + TK) < K;

    // ---- issue next-tile staging (async straight to LDS / regs) ----
    if (hasNext) {
      const int kn = k0 + TK;
      if constexpr (As_k == 1) {
#pragma unroll
        for (int u = 0; u < 2; ++u)
          ASYNC_LDS_B128(lds_off(&Asm[nxt][acr[u]][ack]),
                         (unsigned)(((tileM + acr[u]) * As_m + kn + ack) * 2),
                         A);
      } else {
#pragma unroll
        for (int u = 0; u < 2; ++u)
          pa[u] = *(const v8h*)&A[(kn + tkk[u]) * As_k + (tileM + tc8)];
      }
      if constexpr (Bs_k == 1) {
#pragma unroll
        for (int u = 0; u < 2; ++u)
          ASYNC_LDS_B128(lds_off(&Bsm[nxt][acr[u]][ack]),
                         (unsigned)(((tileN + acr[u]) * Bs_n + kn + ack) * 2),
                         B);
      } else {
#pragma unroll
        for (int u = 0; u < 2; ++u)
          pb[u] = *(const v8h*)&B[(kn + tkk[u]) * Bs_k + (tileN + tc8)];
      }
    }

    // ---- compute current tile (ISA 7.12.2 fragment layouts) ----
    U16x16 af[2], bf[2];
#pragma unroll
    for (int mi = 0; mi < 2; ++mi) {
      int r = wm + 16 * mi + rl;
      af[mi].h[0] = *(const v8h*)&Asm[cur][r][abase];
      af[mi].h[1] = *(const v8h*)&Asm[cur][r][abase + 16];
    }
#pragma unroll
    for (int ni = 0; ni < 2; ++ni) {
      int c = wn + 16 * ni + rl;
      bf[ni].h[0] = *(const v8h*)&Bsm[cur][c][bbase];
      bf[ni].h[1] = *(const v8h*)&Bsm[cur][c][bbase + 16];
    }
#pragma unroll
    for (int mi = 0; mi < 2; ++mi)
#pragma unroll
      for (int ni = 0; ni < 2; ++ni)
        acc[mi][ni] = __builtin_amdgcn_wmma_f32_16x16x32_f16(
            false, af[mi].v, false, bf[ni].v, (short)0, acc[mi][ni], false,
            false);

    // ---- complete next-tile staging ----
    if (hasNext) {
      if constexpr (As_k != 1) {
#pragma unroll
        for (int u = 0; u < 2; ++u)
#pragma unroll
          for (int j = 0; j < 8; ++j) Asm[nxt][tc8 + j][tkk[u]] = pa[u][j];
      }
      if constexpr (Bs_k != 1) {
#pragma unroll
        for (int u = 0; u < 2; ++u)
#pragma unroll
          for (int j = 0; j < 8; ++j) Bsm[nxt][tc8 + j][tkk[u]] = pb[u][j];
      }
      if constexpr (As_k == 1 || Bs_k == 1) WAIT_ASYNC();
    }
    __syncthreads();
    cur = nxt;
  }

#pragma unroll
  for (int mi = 0; mi < 2; ++mi)
#pragma unroll
    for (int ni = 0; ni < 2; ++ni) {
      int c  = tileN + wn + 16 * ni + rl;
      int r0 = tileM + wm + 16 * mi + 8 * hs;
#pragma unroll
      for (int i = 0; i < 8; ++i)
        C[(r0 + i) * Cs_m + c * Cs_n] = (_Float16)(alpha * acc[mi][ni][i]);
    }
}

// ---------------------------------------------------------------------------
// In-place row softmax over 1024 f16 columns (f32 math).  One block per row.
// ---------------------------------------------------------------------------
__global__ __launch_bounds__(256) void softmax_rows(_Float16* __restrict__ SP) {
  __shared__ float red[256];
  _Float16* p = SP + (long)blockIdx.x * NS;
  int tid = threadIdx.x;

  v4h raw = *(const v4h*)(p + 4 * tid);
  float v[4];
  float m = -1e30f;
#pragma unroll
  for (int i = 0; i < 4; ++i) {
    v[i] = (float)raw[i];
    m = fmaxf(m, v[i]);
  }
  red[tid] = m;
  __syncthreads();
  for (int s = 128; s > 0; s >>= 1) {
    if (tid < s) red[tid] = fmaxf(red[tid], red[tid + s]);
    __syncthreads();
  }
  m = red[0];
  __syncthreads();

  float sum = 0.f;
#pragma unroll
  for (int i = 0; i < 4; ++i) {
    v[i] = __expf(v[i] - m);
    sum += v[i];
  }
  red[tid] = sum;
  __syncthreads();
  for (int s = 128; s > 0; s >>= 1) {
    if (tid < s) red[tid] += red[tid + s];
    __syncthreads();
  }
  float inv = 1.0f / red[0];
  v4h o;
#pragma unroll
  for (int i = 0; i < 4; ++i) o[i] = (_Float16)(v[i] * inv);
  *(v4h*)(p + 4 * tid) = o;
}

// ---------------------------------------------------------------------------
// Final fused kernel: out[b,s,d] = -( A1@B1 + A2@B2 )[s,d] + POS_SCALE*pos(s)
// A1/A2 k-contig (async), B1/B2 n-contig (register + LDS scatter).
// Double-buffered like the generic GEMM.  f32 output.
// ---------------------------------------------------------------------------
__global__ __launch_bounds__(NTHR) void final_gemm(
    const _Float16* __restrict__ A1, const _Float16* __restrict__ B1,
    const _Float16* __restrict__ A2, const _Float16* __restrict__ B2,
    float* __restrict__ out) {
  __shared__ alignas(16) _Float16 A1s[2][TM][TK];
  __shared__ alignas(16) _Float16 B1s[2][TN][TK];
  __shared__ alignas(16) _Float16 A2s[2][TM][TK];
  __shared__ alignas(16) _Float16 B2s[2][TN][TK];

  const int b = blockIdx.z;
  A1 += b * (NS * NHZ);
  A2 += b * (NS * NHZ);
  out += (long)b * NS * ND;

  const int tileM = blockIdx.y * TM;
  const int tileN = blockIdx.x * TN;
  const int tid   = threadIdx.x;
  const int wave  = tid >> 5;
  const int lane  = tid & 31;
  const int wm    = (wave >> 1) * 32;
  const int wn    = (wave & 1) * 32;
  const int rl    = lane & 15;
  const int hs    = lane >> 4;
  const int abase = hs * 8;
  const int bbase = hs * 8;

  const int acr[2] = {tid >> 2, (tid + NTHR) >> 2};
  const int ack    = (tid & 3) << 3;
  const int tkk[2] = {tid >> 3, (tid + NTHR) >> 3};
  const int tn8    = (tid & 7) << 3;

  v8f acc[2][2] = {};
  v8h p1[2], p2[2];

  // stage tile 0
#pragma unroll
  for (int u = 0; u < 2; ++u) {
    ASYNC_LDS_B128(lds_off(&A1s[0][acr[u]][ack]),
                   (unsigned)(((tileM + acr[u]) * NHZ + ack) * 2), A1);
    ASYNC_LDS_B128(lds_off(&A2s[0][acr[u]][ack]),
                   (unsigned)(((tileM + acr[u]) * NHZ + ack) * 2), A2);
    v8h v1 = *(const v8h*)&B1[tkk[u] * ND + (tileN + tn8)];
    v8h v2 = *(const v8h*)&B2[tkk[u] * ND + (tileN + tn8)];
#pragma unroll
    for (int j = 0; j < 8; ++j) {
      B1s[0][tn8 + j][tkk[u]] = v1[j];
      B2s[0][tn8 + j][tkk[u]] = v2[j];
    }
  }
  WAIT_ASYNC();
  __syncthreads();

  int cur = 0;
  for (int k0 = 0; k0 < NHZ; k0 += TK) {
    const int nxt      = cur ^ 1;
    const bool hasNext = (k0 + TK) < NHZ;

    if (hasNext) {
      const int kn = k0 + TK;
#pragma unroll
      for (int u = 0; u < 2; ++u) {
        ASYNC_LDS_B128(lds_off(&A1s[nxt][acr[u]][ack]),
                       (unsigned)(((tileM + acr[u]) * NHZ + kn + ack) * 2), A1);
        ASYNC_LDS_B128(lds_off(&A2s[nxt][acr[u]][ack]),
                       (unsigned)(((tileM + acr[u]) * NHZ + kn + ack) * 2), A2);
        p1[u] = *(const v8h*)&B1[(kn + tkk[u]) * ND + (tileN + tn8)];
        p2[u] = *(const v8h*)&B2[(kn + tkk[u]) * ND + (tileN + tn8)];
      }
    }

    U16x16 a1f[2], a2f[2], b1f[2], b2f[2];
#pragma unroll
    for (int mi = 0; mi < 2; ++mi) {
      int r = wm + 16 * mi + rl;
      a1f[mi].h[0] = *(const v8h*)&A1s[cur][r][abase];
      a1f[mi].h[1] = *(const v8h*)&A1s[cur][r][abase + 16];
      a2f[mi].h[0] = *(const v8h*)&A2s[cur][r][abase];
      a2f[mi].h[1] = *(const v8h*)&A2s[cur][r][abase + 16];
    }
#pragma unroll
    for (int ni = 0; ni < 2; ++ni) {
      int c = wn + 16 * ni + rl;
      b1f[ni].h[0] = *(const v8h*)&B1s[cur][c][bbase];
      b1f[ni].h[1] = *(const v8h*)&B1s[cur][c][bbase + 16];
      b2f[ni].h[0] = *(const v8h*)&B2s[cur][c][bbase];
      b2f[ni].h[1] = *(const v8h*)&B2s[cur][c][bbase + 16];
    }
#pragma unroll
    for (int mi = 0; mi < 2; ++mi)
#pragma unroll
      for (int ni = 0; ni < 2; ++ni) {
        acc[mi][ni] = __builtin_amdgcn_wmma_f32_16x16x32_f16(
            false, a1f[mi].v, false, b1f[ni].v, (short)0, acc[mi][ni], false,
            false);
        acc[mi][ni] = __builtin_amdgcn_wmma_f32_16x16x32_f16(
            false, a2f[mi].v, false, b2f[ni].v, (short)0, acc[mi][ni], false,
            false);
      }

    if (hasNext) {
#pragma unroll
      for (int u = 0; u < 2; ++u)
#pragma unroll
        for (int j = 0; j < 8; ++j) {
          B1s[nxt][tn8 + j][tkk[u]] = p1[u][j];
          B2s[nxt][tn8 + j][tkk[u]] = p2[u][j];
        }
      WAIT_ASYNC();
    }
    __syncthreads();
    cur = nxt;
  }

#pragma unroll
  for (int mi = 0; mi < 2; ++mi)
#pragma unroll
    for (int ni = 0; ni < 2; ++ni) {
      int c  = tileN + wn + 16 * ni + rl;
      int r0 = tileM + wm + 16 * mi + 8 * hs;
#pragma unroll
      for (int i = 0; i < 8; ++i) {
        int r = r0 + i;
        float pos = POS_SCALE_F * ((float)r * (1.0f / 1023.0f) - 0.5f);
        out[r * ND + c] = pos - acc[mi][ni][i];
      }
    }
}

// ---------------------------------------------------------------------------
extern "C" void kernel_launch(void* const* d_in, const int* in_sizes, int n_in,
                              void* d_out, int out_size, void* d_ws,
                              size_t ws_size, hipStream_t stream) {
  const float* x  = (const float*)d_in[0];  // [2,1024,1024]
  const float* wq = (const float*)d_in[1];  // [16,64,1024]
  const float* wk = (const float*)d_in[2];  // [16,64,1024]
  float* out = (float*)d_out;               // [2,1024,1024]

  // workspace layout (f16), 88 MB total
  char* ws = (char*)d_ws;
  _Float16* SP  = (_Float16*)ws;               // [B,H,S,S]   64 MB
  _Float16* Qh  = (_Float16*)(ws + 67108864);  // [B,H,S,Z]    4 MB
  _Float16* Kh  = Qh  + 2097152;               // [B,H,S,Z]    4 MB
  _Float16* Abf = Kh  + 2097152;               // [B,S,HZ]     4 MB
  _Float16* Bvf = Abf + 2097152;               // [B,S,HZ]     4 MB
  _Float16* xh  = Bvf + 2097152;               // [B,S,D]      4 MB
  _Float16* wqh = xh  + 2097152;               // [H,Z,D]      2 MB
  _Float16* wkh = wqh + 1048576;               // [H,Z,D]      2 MB

  // 1) converts
  cvt_f32_f16<<<2097152 / 4 / 256, 256, 0, stream>>>(x, xh, 2097152 / 4);
  cvt_f32_f16<<<1048576 / 4 / 256, 256, 0, stream>>>(wq, wqh, 1048576 / 4);
  cvt_f32_f16<<<1048576 / 4 / 256, 256, 0, stream>>>(wk, wkh, 1048576 / 4);

  dim3 blk(NTHR);

  // 2) Q = x @ wq^T : M=S,N=Z,K=D; A k-contig, B k-contig
  gemm_f16<ND, ND, 1, 1, ND, NZ, 1, NS * ND, 0, 0, NZ * ND, NH * NS * NZ,
           NS * NZ, NH>
      <<<dim3(NZ / TN, NS / TM, NB * NH), blk, 0, stream>>>(xh, wqh, Qh, 1.0f);
  // 3) K = x @ wk^T
  gemm_f16<ND, ND, 1, 1, ND, NZ, 1, NS * ND, 0, 0, NZ * ND, NH * NS * NZ,
           NS * NZ, NH>
      <<<dim3(NZ / TN, NS / TM, NB * NH), blk, 0, stream>>>(xh, wkh, Kh, 1.0f);

  // 4) SP = BETA * Q @ K^T : M=S,N=S,K=Z; A k-contig, B k-contig
  gemm_f16<NZ, NZ, 1, 1, NZ, NS, 1, NH * NS * NZ, NS * NZ, NH * NS * NZ,
           NS * NZ, NH * NS * NS, NS * NS, NH>
      <<<dim3(NS / TN, NS / TM, NB * NH), blk, 0, stream>>>(Qh, Kh, SP, BETA_F);

  // 5) P = softmax rows (in place)
  softmax_rows<<<NB * NH * NS, 256, 0, stream>>>(SP);

  // 6) Abf[b,s,hz] = P @ K : M=S,N=Z,K=S; A k-contig, B n-contig
  gemm_f16<NS, NS, 1, NZ, 1, NHZ, 1, NH * NS * NS, NS * NS, NH * NS * NZ,
           NS * NZ, NS * NHZ, NZ, NH>
      <<<dim3(NZ / TN, NS / TM, NB * NH), blk, 0, stream>>>(SP, Kh, Abf, 1.0f);

  // 7) Bvf[b,s,hz] = P^T @ Q : A m-contig (element (m,k) = P[k*S+m]), B n-contig
  gemm_f16<NS, 1, NS, NZ, 1, NHZ, 1, NH * NS * NS, NS * NS, NH * NS * NZ,
           NS * NZ, NS * NHZ, NZ, NH>
      <<<dim3(NZ / TN, NS / TM, NB * NH), blk, 0, stream>>>(SP, Qh, Bvf, 1.0f);

  // 8) out = -(Abf @ wq_flat + Bvf @ wk_flat) + pos
  final_gemm<<<dim3(ND / TN, NS / TM, NB), blk, 0, stream>>>(Abf, wqh, Bvf, wkh,
                                                             out);
}
